// MutualAttentionFusion_27118423507780
// MI455X (gfx1250) — matrix operand order repaired and checked by reading, not compiled, not run on hardware
//
#include <hip/hip_runtime.h>
#include <hip/hip_bf16.h>
#include <math.h>

typedef __bf16 bf16;
typedef __bf16 v8bf  __attribute__((ext_vector_type(8)));
typedef __bf16 v16bf __attribute__((ext_vector_type(16)));
typedef float  v8f   __attribute__((ext_vector_type(8)));
typedef int    v4i   __attribute__((ext_vector_type(4)));

#define KB 4
#define KC 256
#define KN 4096
#define QP 264   // padded channel stride for Q strip in LDS (528B rows, conflict-free)
#define KP 264   // padded channel stride for key strip (col layout)
#define VP 40    // padded key stride for key strip (row layout)

// dynamic LDS partition (halves); no static __shared__ in this kernel, so these
// are also the raw hardware LDS byte offsets (x2).
#define SM_Q  0                          // [128][QP]
#define SM_K  (128 * QP)                 // [2][32][KP]
#define SM_V  (SM_K + 2 * 32 * KP)       // [2][KC][VP]
#define SM_P  (SM_V + 2 * KC * VP)       // [8][16][32]
#define SM_HALVES (SM_P + 8 * 16 * 32)   // 75264 halves = 150528 bytes

#if defined(__has_builtin)
#if __has_builtin(__builtin_amdgcn_global_load_async_to_lds_b128)
#define HAVE_ASYNC_LDS 1
#endif
#endif
#ifndef HAVE_ASYNC_LDS
#define HAVE_ASYNC_LDS 0
#endif

#if HAVE_ASYNC_LDS
// builtin signature (from diagnostics): (int4 AS1*, int4 AS3*, imm int, imm int)
typedef __attribute__((address_space(1))) v4i* gv4i_p;
typedef __attribute__((address_space(3))) v4i* lv4i_p;
#endif

static __device__ __forceinline__ bf16 to_bf16(float x) { return (bf16)x; }

static __device__ __forceinline__ v16bf cat16(v8bf lo, v8bf hi) {
  v16bf r;
#pragma unroll
  for (int e = 0; e < 8; ++e) { r[e] = lo[e]; r[8 + e] = hi[e]; }
  return r;
}

// cooperative stage of one 32-key strip into LDS in both layouts (64B/lane chunks)
static __device__ __forceinline__ void stage_strip(const bf16* __restrict__ Kc,
                                                   const bf16* __restrict__ Vr,
                                                   bf16* __restrict__ smem,
                                                   int jb, int tid,
                                                   unsigned kbase, unsigned vbase) {
  const int k = tid >> 3, seg = tid & 7;
  const bf16* gk = Kc + (size_t)(jb + k) * KC + seg * 32;
  const bf16* gv = Vr + (size_t)tid * KN + jb;
  const unsigned koff = kbase + (unsigned)(k * KP + seg * 32) * 2u;
  const unsigned voff = vbase + (unsigned)(tid * VP) * 2u;
#if HAVE_ASYNC_LDS
  // CDNA5 async copy: no VGPR round-trip, tracked with ASYNCcnt
  gv4i_p ggk = (gv4i_p)(uintptr_t)gk;
  gv4i_p ggv = (gv4i_p)(uintptr_t)gv;
  lv4i_p lk = (lv4i_p)(uintptr_t)koff;
  lv4i_p lv = (lv4i_p)(uintptr_t)voff;
#pragma unroll
  for (int q = 0; q < 4; ++q) {
    __builtin_amdgcn_global_load_async_to_lds_b128(ggk + q, lk + q, 0, 0);
    __builtin_amdgcn_global_load_async_to_lds_b128(ggv + q, lv + q, 0, 0);
  }
#else
  const uint4* g = (const uint4*)gk;
  uint4* s = (uint4*)((char*)smem + koff);
  s[0] = g[0]; s[1] = g[1]; s[2] = g[2]; s[3] = g[3];
  const uint4* g2 = (const uint4*)gv;
  uint4* s2 = (uint4*)((char*)smem + voff);
  s2[0] = g2[0]; s2[1] = g2[1]; s2[2] = g2[2]; s2[3] = g2[3];
#endif
}

// ---------------------------------------------------------------------------
// Kernel 1: f32 -> bf16 conversion into BOTH layouts (coalesced via LDS tile):
//   col layout: X[n][c], row layout: X[c][n]
// ---------------------------------------------------------------------------
__global__ __launch_bounds__(256) void convert_kernel(
    const float* __restrict__ V, const float* __restrict__ T,
    bf16* __restrict__ Vcol, bf16* __restrict__ Tcol,
    bf16* __restrict__ Vrow, bf16* __restrict__ Trow) {
  __shared__ float tile[32][33];
  const int b = blockIdx.z >> 1;
  const bool isT = blockIdx.z & 1;
  const float* src = isT ? T : V;
  bf16* rowd = isT ? Trow : Vrow;
  bf16* cold = isT ? Tcol : Vcol;
  const size_t base = (size_t)b * KC * KN;
  src += base; rowd += base; cold += base;
  const int n0 = blockIdx.x * 32, c0 = blockIdx.y * 32;
  const int tx = threadIdx.x, ty = threadIdx.y;
#pragma unroll
  for (int r = 0; r < 4; ++r) {
    int c = c0 + ty + r * 8;
    float x = src[(size_t)c * KN + n0 + tx];
    rowd[(size_t)c * KN + n0 + tx] = to_bf16(x);
    tile[ty + r * 8][tx] = x;
  }
  __syncthreads();
#pragma unroll
  for (int r = 0; r < 4; ++r) {
    int nl = ty + r * 8;
    cold[(size_t)(n0 + nl) * KC + c0 + tx] = to_bf16(tile[tx][nl]);
  }
}

// ---------------------------------------------------------------------------
// Kernel 2: fused flash-attention pass (dir 0: V attends T; dir 1: T attends V)
// 8 waves x 16 query rows; Q strip resident in LDS; double-buffered key strips
// staged with async global->LDS copies; software-pipelined fragment loads.
// ---------------------------------------------------------------------------
__global__ __launch_bounds__(256) void flash_kernel(
    const bf16* __restrict__ Vcol, const bf16* __restrict__ Tcol,
    const bf16* __restrict__ Vrow, const bf16* __restrict__ Trow,
    const float* __restrict__ Vin, const float* __restrict__ Tin,
    float* __restrict__ out) {
  extern __shared__ __align__(16) bf16 smem[];

  const int b = blockIdx.y;
  const int dir = blockIdx.z;
  const size_t mats = (size_t)KC * KN;
  const bf16* Qcol = (dir == 0 ? Vcol : Tcol) + (size_t)b * mats;
  const bf16* Kc   = (dir == 0 ? Tcol : Vcol) + (size_t)b * mats;
  const bf16* Vr   = (dir == 0 ? Trow : Vrow) + (size_t)b * mats;
  const float* add = (dir == 0 ? Vin : Tin) + (size_t)b * mats;
  float* outp = out + ((size_t)b * 2 * KC + (size_t)dir * KC) * KN;

  const int tid    = threadIdx.x;
  const int wave   = tid >> 5;
  const int lane   = tid & 31;
  const int lane15 = lane & 15;
  const int hi     = lane >> 4;
  const int iblk   = blockIdx.x * 128;

  // stage Q strip (128 x 256 bf16) into LDS, 256B per thread
  {
    const int row = tid >> 1, seg = (tid & 1) * 128;
    const bf16* gq = Qcol + (size_t)(iblk + row) * KC + seg;
    const unsigned qoff = (unsigned)((SM_Q + row * QP + seg) * 2);
#if HAVE_ASYNC_LDS
    gv4i_p gg = (gv4i_p)(uintptr_t)gq;
    lv4i_p lq = (lv4i_p)(uintptr_t)qoff;
#pragma unroll
    for (int q = 0; q < 16; ++q)
      __builtin_amdgcn_global_load_async_to_lds_b128(gg + q, lq + q, 0, 0);
#else
    const uint4* g = (const uint4*)gq;
    uint4* s = (uint4*)((char*)smem + qoff);
#pragma unroll
    for (int q = 0; q < 16; ++q) s[q] = g[q];
#endif
  }
  // stage first key strip into buffer 0
  stage_strip(Kc, Vr, smem, 0, tid, SM_K * 2u, SM_V * 2u);

  const v8f z8 = {0.f, 0.f, 0.f, 0.f, 0.f, 0.f, 0.f, 0.f};
  v8f acc[16];
#pragma unroll
  for (int t = 0; t < 16; ++t) acc[t] = z8;
  float m_run[8], l_run[8];
#pragma unroll
  for (int r = 0; r < 8; ++r) { m_run[r] = -__builtin_inff(); l_run[r] = 0.f; }

  const bf16* qbase = smem + SM_Q + (wave * 16 + lane15) * QP + hi * 16;

  for (int jb = 0; jb < KN; jb += 32) {
    const int cur = (jb >> 5) & 1;
#if HAVE_ASYNC_LDS
    asm volatile("s_wait_asynccnt 0" ::: "memory");  // staged strip landed in LDS
#endif
    __syncthreads();  // staged data for `cur` visible; prev reads of cur^1 done
    if (jb + 32 < KN)  // prefetch next strip into the other buffer
      stage_strip(Kc, Vr, smem, jb + 32, tid,
                  (unsigned)((SM_K + (cur ^ 1) * (32 * KP)) * 2),
                  (unsigned)((SM_V + (cur ^ 1) * (KC * VP)) * 2));

    const bf16* kc0 = smem + SM_K + cur * (32 * KP) + lane15 * KP + hi * 16;
    const bf16* kc1 = kc0 + 16 * KP;
    const bf16* vcur = smem + SM_V + cur * (KC * VP) + lane15 * VP + hi * 16;

    // S tiles: 16 queries x 32 keys, K = 256 channels; pipelined fragment loads
    v8f s0 = z8, s1 = z8;
    v16bf aq = cat16(*(const v8bf*)qbase, *(const v8bf*)(qbase + 8));
    v16bf b0 = *(const v16bf*)kc0;
    v16bf b1 = *(const v16bf*)kc1;
#pragma unroll
    for (int kk = 0; kk < 8; ++kk) {
      v16bf aqn, b0n, b1n;
      if (kk < 7) {
        const bf16* qp = qbase + (kk + 1) * 32;
        aqn = cat16(*(const v8bf*)qp, *(const v8bf*)(qp + 8));
        b0n = *(const v16bf*)(kc0 + (kk + 1) * 32);
        b1n = *(const v16bf*)(kc1 + (kk + 1) * 32);
      }
      s0 = __builtin_amdgcn_wmma_f32_16x16x32_bf16(false, aq, false, b0,
                                                   (short)0, s0, false, false);
      s1 = __builtin_amdgcn_wmma_f32_16x16x32_bf16(false, aq, false, b1,
                                                   (short)0, s1, false, false);
      if (kk < 7) { aq = aqn; b0 = b0n; b1 = b1n; }
    }

    // ---- online softmax over the 32 new columns (rows live per half-wave) ----
    float mn[8];
#pragma unroll
    for (int r = 0; r < 8; ++r) mn[r] = fmaxf(s0[r], s1[r]);
#pragma unroll
    for (int msk = 1; msk <= 8; msk <<= 1)
#pragma unroll
      for (int r = 0; r < 8; ++r) mn[r] = fmaxf(mn[r], __shfl_xor(mn[r], msk, 32));

    float rs[8];
#pragma unroll
    for (int r = 0; r < 8; ++r) {          // exp in place; mn becomes the rescale
      const float mi = fmaxf(m_run[r], mn[r]);
      const float sc = __expf(m_run[r] - mi);
      m_run[r] = mi;
      s0[r] = __expf(s0[r] - mi);
      s1[r] = __expf(s1[r] - mi);
      rs[r] = s0[r] + s1[r];
      l_run[r] *= sc;
      mn[r] = sc;
    }
#pragma unroll
    for (int msk = 1; msk <= 8; msk <<= 1)
#pragma unroll
      for (int r = 0; r < 8; ++r) rs[r] += __shfl_xor(rs[r], msk, 32);
#pragma unroll
    for (int r = 0; r < 8; ++r) l_run[r] += rs[r];
#pragma unroll
    for (int t = 0; t < 16; ++t)
#pragma unroll
      for (int r = 0; r < 8; ++r) acc[t][r] *= mn[r];

    // ---- P: D-tile layout -> A-fragment layout via per-wave LDS slab ----
    bf16* pslab = smem + SM_P + wave * (16 * 32);
#pragma unroll
    for (int r = 0; r < 8; ++r) {
      pslab[(r + 8 * hi) * 32 + lane15]      = to_bf16(s0[r]);
      pslab[(r + 8 * hi) * 32 + lane15 + 16] = to_bf16(s1[r]);
    }
    // intra-wave only: CDNA5 split DS counter wait (also a compiler mem barrier)
    asm volatile("s_wait_dscnt 0" ::: "memory");
    const bf16* pp = pslab + lane15 * 32 + hi * 16;
    v16bf pa = cat16(*(const v8bf*)pp, *(const v8bf*)(pp + 8));

    // ---- O += P (16x32) x Val (32 keys x 16 channels); pipelined B loads ----
    v16bf vb = *(const v16bf*)vcur;
#pragma unroll
    for (int t = 0; t < 16; ++t) {
      v16bf vbn;
      if (t < 15) vbn = *(const v16bf*)(vcur + (t + 1) * (16 * VP));
      acc[t] = __builtin_amdgcn_wmma_f32_16x16x32_bf16(false, pa, false, vb,
                                                       (short)0, acc[t], false, false);
      if (t < 15) vb = vbn;
    }
  }

  // epilogue: out[b, dir*C + c, i] = add[c, i] + acc/l
#pragma unroll
  for (int t = 0; t < 16; ++t) {
#pragma unroll
    for (int r = 0; r < 8; ++r) {
      const int i = iblk + wave * 16 + r + 8 * hi;
      const int c = t * 16 + lane15;
      const float o = acc[t][r] / l_run[r];
      outp[(size_t)c * KN + i] = add[(size_t)c * KN + i] + o;
    }
  }
}

extern "C" void kernel_launch(void* const* d_in, const int* in_sizes, int n_in,
                              void* d_out, int out_size, void* d_ws, size_t ws_size,
                              hipStream_t stream) {
  (void)in_sizes; (void)n_in; (void)out_size; (void)ws_size;
  const float* V = (const float*)d_in[0];
  const float* T = (const float*)d_in[1];
  float* out = (float*)d_out;

  bf16* ws = (bf16*)d_ws;
  const size_t per = (size_t)KB * KC * KN;   // elements per staged matrix
  bf16* Vcol = ws;
  bf16* Tcol = ws + per;
  bf16* Vrow = ws + 2 * per;
  bf16* Trow = ws + 3 * per;                 // 32 MB of workspace total

  convert_kernel<<<dim3(KN / 32, KC / 32, 2 * KB), dim3(32, 8), 0, stream>>>(
      V, T, Vcol, Tcol, Vrow, Trow);
  flash_kernel<<<dim3(KN / 128, KB, 2), dim3(256), (unsigned)(SM_HALVES * 2),
                 stream>>>(Vcol, Tcol, Vrow, Trow, V, T, out);
}
